// V28Model_43250320670960
// MI455X (gfx1250) — matrix operand order repaired and checked by reading, not compiled, not run on hardware
//
#include <hip/hip_runtime.h>
#include <cstddef>
#include <cstdint>

// ---------------- model constants ----------------
static constexpr int Vv   = 50280;
static constexpr int Dd   = 768;
static constexpr int Ee   = 1536;
static constexpr int E2   = 3072;   // 2*E
static constexpr int Ss   = 16;
static constexpr int Kc   = 4;
static constexpr int Rr   = 48;
static constexpr int RP   = 64;     // dt K padded to mult of 32
static constexpr int XP   = 80;     // R + 2S
static constexpr int NLl  = 8;
static constexpr int SEQ  = 512;
static constexpr int LORA_R = 8;
static constexpr float LORA_SCALE = 2.0f;
static constexpr float NEGV = -1e30f;

typedef __attribute__((ext_vector_type(16))) __bf16 v16bf;
typedef __attribute__((ext_vector_type(8)))  float  v8f;
typedef __attribute__((ext_vector_type(4)))  int    i4;

// Fragment alias: two 128-bit loads fill the 8-VGPR WMMA operand directly.
union Frag {
  v16bf f;
  i4    i[2];
};

// ================= WMMA GEMM =================
// Out[M,N] (f32) = A[M,K](bf16, row-major ld=K) * W[N,K](bf16, row-major ld=K)^T
// EPI: 0 = store, 1 = accumulate into Out, 2 = +bias then softplus (dt path)
// NT:  16-col tiles per wave (grid chosen so coverage is EXACT -> branch-free).
// grid: (M/128, N/(NT*16)), block 256 (8 waves); wave w -> rows [bx*128+w*16, +16).
// Double-buffered k-loop: two disjoint fragment sets so WMMA consumes loads
// issued one iteration earlier (partial waits, real load/compute overlap).
// NOTE: all K dims used (64, 768, 1536) have an even number of 32-steps.
template <int EPI, int NT>
__global__ __launch_bounds__(256) void gemm_bf16_wmma(
    const __bf16* __restrict__ A, const __bf16* __restrict__ W,
    float* __restrict__ Out, const float* __restrict__ bias,
    int N, int Kd) {
  const int lane  = threadIdx.x & 31;
  const int wave  = threadIdx.x >> 5;
  const int mBase = blockIdx.x * 128 + wave * 16;
  const int nBase = blockIdx.y * (NT * 16);
  const int half  = lane >> 4;   // K-half selector
  const int l15   = lane & 15;

  v8f acc[NT];
#pragma unroll
  for (int t = 0; t < NT; ++t)
#pragma unroll
    for (int i = 0; i < 8; ++i) acc[t][i] = 0.0f;

  // A fragment rows: lane(0-15)=row, interleaved K halves (+0/+8, +16/+24)
  const __bf16* aRow  = A + (size_t)(mBase + l15) * Kd + half * 8;
  // B fragment: lane = output column, interleaved K halves (+0/+8, +16/+24)
  const __bf16* wRow0 = W + (size_t)(nBase + l15) * Kd + half * 8;

  Frag a0, a1, b0[NT], b1[NT];

  auto loadF = [&](int k, Frag& a, Frag (&b)[NT]) {
    a.i[0] = *(const i4*)(aRow + k);        // K k..k+7   | k+8..k+15
    a.i[1] = *(const i4*)(aRow + k + 16);   // K k+16..23 | k+24..31
#pragma unroll
    for (int t = 0; t < NT; ++t) {
      const __bf16* wr = wRow0 + (size_t)t * 16 * Kd + k;
      b[t].i[0] = *(const i4*)(wr);
      b[t].i[1] = *(const i4*)(wr + 16);
    }
  };
  auto chain = [&](const Frag& a, const Frag (&b)[NT]) {
#pragma unroll
    for (int t = 0; t < NT; ++t)
      acc[t] = __builtin_amdgcn_wmma_f32_16x16x32_bf16(
          false, a.f, false, b[t].f, (short)0, acc[t], false, false);
  };

  loadF(0, a0, b0);
  int k = 32;
  for (; k < Kd - 32; k += 64) {
    __builtin_prefetch(aRow + k + 96, 0, 3);  // near-scope prefetch ahead
    loadF(k, a1, b1);
    chain(a0, b0);
    loadF(k + 32, a0, b0);
    chain(a1, b1);
  }
  // k == Kd - 32 here (even step count)
  loadF(k, a1, b1);
  chain(a0, b0);
  chain(a1, b1);

#pragma unroll
  for (int t = 0; t < NT; ++t) {
    const int col = nBase + t * 16 + l15;
#pragma unroll
    for (int r = 0; r < 8; ++r) {
      const int row = mBase + half * 8 + r;  // C/D: VGPR r -> M=r (lanes<16) / M=8+r
      const size_t idx = (size_t)row * N + col;
      float vv = acc[t][r];
      if (EPI == 1) {
        Out[idx] += vv;
      } else if (EPI == 2) {
        float u = vv + bias[col];
        Out[idx] = (u > 20.0f) ? u : log1pf(__expf(u));  // softplus
      } else {
        Out[idx] = vv;
      }
    }
  }
}

// ================= elementwise / small kernels =================

__global__ __launch_bounds__(256) void embed_k(const int* __restrict__ ids,
                                               const float* __restrict__ emb,
                                               float* __restrict__ x) {
  const int l = blockIdx.x;
  const size_t base = (size_t)ids[l] * Dd;
  for (int d = threadIdx.x; d < Dd; d += 256) x[(size_t)l * Dd + d] = emb[base + d];
}

// RMSNorm worker. Row l = blockIdx.x + row0. r = x + (res_in? res_in : 0);
// optional res_out (f32), out_bf (bf16), out_f (f32). Output base = blockIdx.x*D.
__global__ __launch_bounds__(256) void rms_k(
    const float* __restrict__ x, const float* __restrict__ res_in,
    const float* __restrict__ w, float* __restrict__ res_out,
    __bf16* __restrict__ out_bf, float* __restrict__ out_f, int row0) {
  __shared__ float red[8];
  const int l = blockIdx.x + row0;
  const float* xr = x + (size_t)l * Dd;
  const float* rr = res_in ? res_in + (size_t)l * Dd : nullptr;
  float vals[3];
  float ss = 0.0f;
  int i = 0;
  for (int d = threadIdx.x; d < Dd; d += 256, ++i) {
    float v = xr[d] + (rr ? rr[d] : 0.0f);
    vals[i] = v;
    ss += v * v;
    if (res_out) res_out[(size_t)l * Dd + d] = v;
  }
  for (int off = 16; off; off >>= 1) ss += __shfl_down(ss, off);
  if ((threadIdx.x & 31) == 0) red[threadIdx.x >> 5] = ss;
  __syncthreads();
  float tot = 0.0f;
  for (int j = 0; j < 8; ++j) tot += red[j];
  const float scale = rsqrtf(tot / (float)Dd + 1e-5f);
  const size_t ob = (size_t)blockIdx.x * Dd;
  i = 0;
  for (int d = threadIdx.x; d < Dd; d += 256, ++i) {
    float o = vals[i] * scale * w[d];
    if (out_bf) out_bf[ob + d] = (__bf16)o;
    if (out_f)  out_f[ob + d]  = o;
  }
}

// causal depthwise conv(K=4) + bias + SiLU over xi = xz[:, :E]; outputs f32 + bf16
__global__ __launch_bounds__(256) void conv_silu_k(
    const float* __restrict__ xz, const float* __restrict__ cw,
    const float* __restrict__ cb, float* __restrict__ xc,
    __bf16* __restrict__ xc_bf) {
  const int idx = blockIdx.x * 256 + threadIdx.x;
  if (idx >= SEQ * Ee) return;
  const int l = idx / Ee, e = idx - l * Ee;
  float acc = cb[e];
#pragma unroll
  for (int k = 0; k < Kc; ++k) {
    const int ls = l + k - (Kc - 1);
    if (ls >= 0) acc += xz[(size_t)ls * E2 + e] * cw[e * Kc + k];
  }
  const float s = acc / (1.0f + __expf(-acc));
  xc[idx] = s;
  xc_bf[idx] = (__bf16)s;
}

// split dbl (SEQ x 80) into padded-bf16 dt input (SEQ x 64) and B/C (SEQ x 16)
__global__ void dtprep_k(const float* __restrict__ dbl, __bf16* __restrict__ dtin,
                         float* __restrict__ Bm, float* __restrict__ Cm) {
  const int l = blockIdx.x, t = threadIdx.x;  // 64 threads
  const float v = (t < Rr) ? dbl[(size_t)l * XP + t] : 0.0f;
  dtin[(size_t)l * RP + t] = (__bf16)v;
  if (t < Ss) {
    Bm[l * Ss + t] = dbl[(size_t)l * XP + Rr + t];
    Cm[l * Ss + t] = dbl[(size_t)l * XP + Rr + Ss + t];
  }
}

// sequential selective scan; thread = channel e; h[16], A[16] register-resident
__global__ __launch_bounds__(256) void scan_k(
    const float* __restrict__ dt, const float* __restrict__ Bm,
    const float* __restrict__ Cm, const float* __restrict__ xc,
    const float* __restrict__ xz, const float* __restrict__ A_log,
    const float* __restrict__ Dv, __bf16* __restrict__ y_bf) {
  const int e = blockIdx.x * 256 + threadIdx.x;
  float A[Ss], h[Ss];
#pragma unroll
  for (int s = 0; s < Ss; ++s) { A[s] = -__expf(A_log[(size_t)e * Ss + s]); h[s] = 0.0f; }
  const float dve = Dv[e];
  for (int l = 0; l < SEQ; ++l) {
    const float d  = dt[(size_t)l * Ee + e];
    const float xv = xc[(size_t)l * Ee + e];
    const float dx = d * xv;
    float y = 0.0f;
#pragma unroll
    for (int s = 0; s < Ss; ++s) {
      h[s] = h[s] * __expf(d * A[s]) + dx * Bm[l * Ss + s];
      y += h[s] * Cm[l * Ss + s];
    }
    y += dve * xv;
    const float z = xz[(size_t)l * E2 + Ee + e];
    y *= z / (1.0f + __expf(-z));
    y_bf[(size_t)l * Ee + e] = (__bf16)y;
  }
}

__global__ __launch_bounds__(256) void addrow_k(float* __restrict__ x,
                                                const float* __restrict__ w) {
  for (int d = threadIdx.x; d < Dd; d += 256) x[(size_t)blockIdx.x * Dd + d] += w[d];
}

__global__ void f2bf_k(const float* __restrict__ x, __bf16* __restrict__ o, int n) {
  const int i = blockIdx.x * 256 + threadIdx.x;
  if (i < n) o[i] = (__bf16)x[i];
}

__global__ void maskinit_k(float* __restrict__ m) {
  const int i = blockIdx.x * 256 + threadIdx.x;
  if (i < Vv) m[i] = NEGV;
}
__global__ void maskset_k(float* __restrict__ m, const int* __restrict__ ids) {
  const int i = blockIdx.x * 256 + threadIdx.x;
  if (i < SEQ) m[ids[i]] = 0.0f;
}

// wave-per-vocab-row matvec: out[v] = dot(t, lm_w[v]) + mask[v]
__global__ __launch_bounds__(256) void logits_k(
    const float* __restrict__ t, const float* __restrict__ lm,
    const float* __restrict__ mask, float* __restrict__ out) {
  const int v = blockIdx.x * 8 + (threadIdx.x >> 5);
  if (v >= Vv) return;
  const int lane = threadIdx.x & 31;
  const float* r = lm + (size_t)v * Dd;
  float s = 0.0f;
  for (int d = lane; d < Dd; d += 32) s += r[d] * t[d];
  for (int off = 16; off; off >>= 1) s += __shfl_down(s, off);
  if (lane == 0) out[v] = s + mask[v];
}

// weight -> bf16, optionally fusing rank-8 LoRA: w + sc * (lB @ lA)
__global__ void wfuse_k(const float* __restrict__ w, const float* __restrict__ lA,
                        const float* __restrict__ lB, __bf16* __restrict__ out,
                        int Nn, int Kk, float sc) {
  const size_t i = (size_t)blockIdx.x * 256 + threadIdx.x;
  if (i >= (size_t)Nn * Kk) return;
  const int n = (int)(i / Kk), k = (int)(i - (size_t)n * Kk);
  float v = w[i];
  if (lA) {
    float a = 0.0f;
#pragma unroll
    for (int r = 0; r < LORA_R; ++r) a += lB[n * LORA_R + r] * lA[(size_t)r * Kk + k];
    v += sc * a;
  }
  out[i] = (__bf16)v;
}

// dt_w (E x 48) -> bf16 (E x 64) zero-padded
__global__ void wdtpad_k(const float* __restrict__ w, __bf16* __restrict__ out) {
  const size_t i = (size_t)blockIdx.x * 256 + threadIdx.x;
  if (i >= (size_t)Ee * RP) return;
  const int e = (int)(i / RP), r = (int)(i - (size_t)e * RP);
  out[i] = (__bf16)((r < Rr) ? w[(size_t)e * Rr + r] : 0.0f);
}

// ================= host orchestration =================
extern "C" void kernel_launch(void* const* d_in, const int* in_sizes, int n_in,
                              void* d_out, int out_size, void* d_ws, size_t ws_size,
                              hipStream_t stream) {
  (void)in_sizes; (void)n_in;
  // ---- inputs (setup_inputs order) ----
  const int*   ids     = (const int*)d_in[0];
  const float* emb     = (const float*)d_in[2];
  const float* lm_w    = (const float*)d_in[3];
  const float* ln_w    = (const float*)d_in[4];
  const float* in_w    = (const float*)d_in[5];
  const float* conv_w  = (const float*)d_in[6];
  const float* conv_b  = (const float*)d_in[7];
  const float* xproj_w = (const float*)d_in[8];
  const float* dt_w    = (const float*)d_in[9];
  const float* dt_b    = (const float*)d_in[10];
  const float* A_log   = (const float*)d_in[11];
  const float* Dp      = (const float*)d_in[12];
  const float* out_w   = (const float*)d_in[13];
  const float* lA_in   = (const float*)d_in[14];
  const float* lB_in   = (const float*)d_in[15];
  const float* lA_out  = (const float*)d_in[16];
  const float* lB_out  = (const float*)d_in[17];
  const float* step_w  = (const float*)d_in[18];
  const float* loop_w  = (const float*)d_in[19];
  const float* normf_w = (const float*)d_in[20];
  const float* m3_in_w    = (const float*)d_in[21];
  const float* m3_conv_w  = (const float*)d_in[22];
  const float* m3_conv_b  = (const float*)d_in[23];
  const float* m3_xproj_w = (const float*)d_in[24];
  const float* m3_dt_w    = (const float*)d_in[25];
  const float* m3_dt_b    = (const float*)d_in[26];
  const float* m3_A_log   = (const float*)d_in[27];
  const float* m3_D       = (const float*)d_in[28];
  const float* m3_out_w   = (const float*)d_in[29];
  float* out = (float*)d_out;
  const int n_loops = out_size / Vv;  // host-derivable (device scalar unreadable in capture)

  // ---- workspace bump allocator ----
  char* p = (char*)d_ws;
  auto alloc = [&](size_t bytes) -> void* {
    void* r = (void*)p;
    p += (bytes + 255) & ~(size_t)255;
    return r;
  };
  const size_t szInW  = (size_t)E2 * Dd;  // 3072*768
  const size_t szOutW = (size_t)Dd * Ee;  // 768*1536
  const size_t szXpW  = (size_t)XP * Ee;  // 80*1536
  const size_t szDtW  = (size_t)Ee * RP;  // 1536*64

  __bf16* w_in_bf  = (__bf16*)alloc(NLl * szInW  * 2);
  __bf16* w_out_bf = (__bf16*)alloc(NLl * szOutW * 2);
  __bf16* w_xp_bf  = (__bf16*)alloc(NLl * szXpW  * 2);
  __bf16* w_dt_bf  = (__bf16*)alloc(NLl * szDtW  * 2);
  __bf16* m3_in_bf  = (__bf16*)alloc(szInW  * 2);
  __bf16* m3_out_bf = (__bf16*)alloc(szOutW * 2);
  __bf16* m3_xp_bf  = (__bf16*)alloc(szXpW  * 2);
  __bf16* m3_dt_bf  = (__bf16*)alloc(szDtW  * 2);

  float*  x      = (float*)alloc((size_t)SEQ * Dd * 4);
  float*  res    = (float*)alloc((size_t)SEQ * Dd * 4);
  __bf16* h_bf   = (__bf16*)alloc((size_t)SEQ * Dd * 2);
  float*  xz     = (float*)alloc((size_t)SEQ * E2 * 4);
  float*  xc     = (float*)alloc((size_t)SEQ * Ee * 4);
  __bf16* xc_bf  = (__bf16*)alloc((size_t)SEQ * Ee * 2);
  float*  dbl    = (float*)alloc((size_t)SEQ * XP * 4);
  __bf16* dt_in  = (__bf16*)alloc((size_t)SEQ * RP * 2);
  float*  dtf    = (float*)alloc((size_t)SEQ * Ee * 4);
  float*  Bm     = (float*)alloc((size_t)SEQ * Ss * 4);
  float*  Cm     = (float*)alloc((size_t)SEQ * Ss * 4);
  __bf16* y_bf   = (__bf16*)alloc((size_t)SEQ * Ee * 2);
  float*  mask   = (float*)alloc((size_t)Vv * 4);
  float*  tvec   = (float*)alloc((size_t)Dd * 4);
  if ((size_t)(p - (char*)d_ws) > ws_size) return;  // insufficient scratch

  // ---- one-time weight conversion (deterministic, re-run each call) ----
  auto cdiv = [](size_t a, size_t b) { return (unsigned)((a + b - 1) / b); };
  for (int i = 0; i < NLl; ++i) {
    const int j = i - 6;  // BASE_SPLIT
    const float* lAi = (i >= 6) ? lA_in  + (size_t)j * LORA_R * Dd : nullptr;
    const float* lBi = (i >= 6) ? lB_in  + (size_t)j * E2 * LORA_R : nullptr;
    const float* lAo = (i >= 6) ? lA_out + (size_t)j * LORA_R * Ee : nullptr;
    const float* lBo = (i >= 6) ? lB_out + (size_t)j * Dd * LORA_R : nullptr;
    wfuse_k<<<cdiv(szInW, 256), 256, 0, stream>>>(in_w + i * szInW, lAi, lBi,
                                                  w_in_bf + i * szInW, E2, Dd, LORA_SCALE);
    wfuse_k<<<cdiv(szOutW, 256), 256, 0, stream>>>(out_w + i * szOutW, lAo, lBo,
                                                   w_out_bf + i * szOutW, Dd, Ee, LORA_SCALE);
    wfuse_k<<<cdiv(szXpW, 256), 256, 0, stream>>>(xproj_w + i * szXpW, nullptr, nullptr,
                                                  w_xp_bf + i * szXpW, XP, Ee, 0.0f);
    wdtpad_k<<<cdiv(szDtW, 256), 256, 0, stream>>>(dt_w + (size_t)i * Ee * Rr,
                                                   w_dt_bf + i * szDtW);
  }
  wfuse_k<<<cdiv(szInW, 256), 256, 0, stream>>>(m3_in_w, nullptr, nullptr, m3_in_bf, E2, Dd, 0.0f);
  wfuse_k<<<cdiv(szOutW, 256), 256, 0, stream>>>(m3_out_w, nullptr, nullptr, m3_out_bf, Dd, Ee, 0.0f);
  wfuse_k<<<cdiv(szXpW, 256), 256, 0, stream>>>(m3_xproj_w, nullptr, nullptr, m3_xp_bf, XP, Ee, 0.0f);
  wdtpad_k<<<cdiv(szDtW, 256), 256, 0, stream>>>(m3_dt_w, m3_dt_bf);

  // ---- mamba pass: hbf (SEQ x D bf16) -> outx (SEQ x D f32) ----
  auto mamba = [&](const __bf16* hbf, const __bf16* wi, const float* cw, const float* cb,
                   const __bf16* wxp, const __bf16* wdt, const float* dtb,
                   const float* alog, const float* dv, const __bf16* wo,
                   float* outx, int accum) {
    // xz = h @ in_w^T : N=3072, K=768 (exact 4x64-wide tiling)
    gemm_bf16_wmma<0, 4><<<dim3(SEQ / 128, E2 / 64), 256, 0, stream>>>(hbf, wi, xz, nullptr,
                                                                       E2, Dd);
    conv_silu_k<<<cdiv((size_t)SEQ * Ee, 256), 256, 0, stream>>>(xz, cw, cb, xc, xc_bf);
    // dbl = xc @ xproj^T : N=80 -> NT=1, 5 column blocks (exact)
    gemm_bf16_wmma<0, 1><<<dim3(SEQ / 128, XP / 16), 256, 0, stream>>>(xc_bf, wxp, dbl, nullptr,
                                                                       XP, Ee);
    dtprep_k<<<SEQ, 64, 0, stream>>>(dbl, dt_in, Bm, Cm);
    // dt = softplus(dtin @ dt_w^T + dt_b) : N=1536, K=64
    gemm_bf16_wmma<2, 4><<<dim3(SEQ / 128, Ee / 64), 256, 0, stream>>>(dt_in, wdt, dtf, dtb,
                                                                       Ee, RP);
    scan_k<<<Ee / 256, 256, 0, stream>>>(dtf, Bm, Cm, xc, xz, alog, dv, y_bf);
    // out = y @ out_w^T : N=768, K=1536
    if (accum)
      gemm_bf16_wmma<1, 4><<<dim3(SEQ / 128, Dd / 64), 256, 0, stream>>>(y_bf, wo, outx,
                                                                         nullptr, Dd, Ee);
    else
      gemm_bf16_wmma<0, 4><<<dim3(SEQ / 128, Dd / 64), 256, 0, stream>>>(y_bf, wo, outx,
                                                                         nullptr, Dd, Ee);
  };

  auto block = [&](int i, bool first) {
    // res = x (+ res); h_bf = rms(res)*ln_w
    rms_k<<<SEQ, 256, 0, stream>>>(x, first ? nullptr : res, ln_w + (size_t)i * Dd,
                                   res, h_bf, nullptr, 0);
    mamba(h_bf, w_in_bf + i * szInW, conv_w + (size_t)i * Ee * Kc, conv_b + (size_t)i * Ee,
          w_xp_bf + i * szXpW, w_dt_bf + i * szDtW, dt_b + (size_t)i * Ee,
          A_log + (size_t)i * Ee * Ss, Dp + (size_t)i * Ee, w_out_bf + i * szOutW, x, 0);
  };

  // ---- forward ----
  embed_k<<<SEQ, 256, 0, stream>>>(ids, emb, x);
  for (int i = 0; i < NLl; ++i) block(i, i == 0);

  maskinit_k<<<cdiv(Vv, 256), 256, 0, stream>>>(mask);
  maskset_k<<<cdiv(SEQ, 256), 256, 0, stream>>>(mask, ids);

  for (int li = 0; li < n_loops; ++li) {
    const int sw = (li < 5) ? li : 5;
    addrow_k<<<SEQ, 256, 0, stream>>>(x, step_w + (size_t)sw * Dd);
    block(6, false);
    block(7, false);
    // x = x + mamba3(x)
    f2bf_k<<<cdiv((size_t)SEQ * Dd, 256), 256, 0, stream>>>(x, h_bf, SEQ * Dd);
    mamba(h_bf, m3_in_bf, m3_conv_w, m3_conv_b, m3_xp_bf, m3_dt_bf, m3_dt_b,
          m3_A_log, m3_D, m3_out_bf, x, 1);
    // x = rms(x, loop_w)   (in-place)
    rms_k<<<SEQ, 256, 0, stream>>>(x, nullptr, loop_w, nullptr, nullptr, x, 0);
    // t = rms(x[last] + res[last], normf_w); logits = t . lm_w + mask
    rms_k<<<1, 256, 0, stream>>>(x, res, normf_w, nullptr, nullptr, tvec, SEQ - 1);
    logits_k<<<Vv / 8, 256, 0, stream>>>(tvec, lm_w, mask, out + (size_t)li * Vv);
  }
}
